// ChildSumTreeLSTM_31636729103180
// MI455X (gfx1250) — matrix operand order repaired and checked by reading, not compiled, not run on hardware
//
#include <hip/hip_runtime.h>
#include <hip/hip_bf16.h>

typedef __attribute__((ext_vector_type(16))) __bf16 v16bf;
typedef __attribute__((ext_vector_type(8)))  __bf16 v8bf;
typedef __attribute__((ext_vector_type(8)))  float  v8f;

#define NND  8192
#define FD   512
#define HD   512
#define KCH  8
#define NCOL 2048            // 512 (f-gate cols) + 1536 (iou cols)

#define G    128             // persistent blocks (16 col-slices x 8 replicas)
#define REP  (G / 16)
#define BLSTR 520            // LDS col stride (bf16) -> +4 banks per col, conflict-free b128
#define ALSTR 520

union BF16x16 { v16bf v; v8bf h[2]; };

__device__ __forceinline__ float sigm(float x){ return 1.f/(1.f + __expf(-x)); }
__device__ __forceinline__ float ftanh(float x){ float e = __expf(2.f*x); return (e-1.f)/(e+1.f); }

__device__ __forceinline__ __bf16 f2bf(float f){
  unsigned u = __float_as_uint(f);
  unsigned short s = (unsigned short)((u + 0x7FFFu + ((u >> 16) & 1u)) >> 16); // RNE
  return __builtin_bit_cast(__bf16, s);
}

// grid-wide sense-reversing barrier (all G blocks resident)
__device__ __forceinline__ void gsync(int* cnt, int* gen){
  __threadfence();
  __syncthreads();
  if (threadIdx.x == 0){
    int g = __hip_atomic_load(gen, __ATOMIC_RELAXED, __HIP_MEMORY_SCOPE_AGENT);
    int a = __hip_atomic_fetch_add(cnt, 1, __ATOMIC_ACQ_REL, __HIP_MEMORY_SCOPE_AGENT);
    if (a == G - 1){
      __hip_atomic_store(cnt, 0, __ATOMIC_RELAXED, __HIP_MEMORY_SCOPE_AGENT);
      __hip_atomic_fetch_add(gen, 1, __ATOMIC_RELEASE, __HIP_MEMORY_SCOPE_AGENT);
    } else {
      while (__hip_atomic_load(gen, __ATOMIC_ACQUIRE, __HIP_MEMORY_SCOPE_AGENT) == g)
        __builtin_amdgcn_s_sleep(2);
    }
  }
  __syncthreads();
  __builtin_amdgcn_fence(__ATOMIC_ACQUIRE, "agent");
}

// ---------------- prep kernels ----------------
__global__ void k_cvt_x(const float* __restrict__ x, __bf16* __restrict__ xb, int n){
  int i = blockIdx.x * blockDim.x + threadIdx.x;
  if (i < n) xb[i] = f2bf(x[i]);
}

// dst[col][k] ([2048][512] bf16): col<512 -> row col of Wa; else row col-512 of Wb
__global__ void k_prep_w(const float* __restrict__ Wa, const float* __restrict__ Wb,
                         __bf16* __restrict__ dst){
  int i = blockIdx.x * blockDim.x + threadIdx.x;
  if (i >= NCOL * FD) return;
  int n = i >> 9, k = i & 511;
  int na = (n < HD) ? n : 0;              // clamped: both arms always in-bounds
  int nb = (n >= HD) ? (n - HD) : 0;
  float v = (n < HD) ? Wa[na * FD + k] : Wb[nb * FD + k];
  dst[i] = f2bf(v);
}

__global__ void k_zero_sched(int* __restrict__ sched){
  if (threadIdx.x < 4) sched[threadIdx.x] = 0;   // barCnt, barGen, cnt0, cnt1
}

__global__ void k_init_leaves(const int* __restrict__ children, int* __restrict__ sched){
  int* done  = sched + 8;
  int* list0 = done + NND;
  int i = blockIdx.x * blockDim.x + threadIdx.x;
  if (i >= NND) return;
  done[i] = 0;
  bool leaf = true;
#pragma unroll
  for (int r = 0; r < KCH; ++r) if (children[i * KCH + r] >= 0) leaf = false;
  if (leaf){
    int idx = __hip_atomic_fetch_add(&sched[2], 1, __ATOMIC_RELAXED, __HIP_MEMORY_SCOPE_AGENT);
    list0[idx] = i;
  }
}

// ---------------- phase 1: WX[8192,2048] = Xb @ Wxt^T (bf16 WMMA, f32 acc) ----------------
__global__ __launch_bounds__(256) void k_gemm_wx(const __bf16* __restrict__ Xb,
                                                 const __bf16* __restrict__ Wxt,
                                                 float* __restrict__ WX){
  const int tid  = threadIdx.x;
  const int lane = tid & 31, wv = tid >> 5;
  const int half = lane >> 4, l16 = lane & 15;
  const int job  = blockIdx.x * 8 + wv;
  const int rt   = job >> 5;
  const int cb0  = (job & 31) * 64;
  const int aoff = half * 8;

  const __bf16* aBase = Xb + (size_t)(rt * 16 + l16) * FD;
  const __bf16* bBase[4];
#pragma unroll
  for (int t = 0; t < 4; ++t)
    bBase[t] = Wxt + (size_t)(cb0 + t * 16 + l16) * FD + half * 8;

  v8f acc[4];
#pragma unroll
  for (int t = 0; t < 4; ++t) acc[t] = (v8f){0,0,0,0,0,0,0,0};

#pragma unroll 4
  for (int ks = 0; ks < 16; ++ks){
    const int k0 = ks * 32;
    BF16x16 a;
    a.h[0] = *(const v8bf*)(aBase + k0 + aoff);
    a.h[1] = *(const v8bf*)(aBase + k0 + 16 + aoff);
#pragma unroll
    for (int t = 0; t < 4; ++t){
      BF16x16 b;
      b.h[0] = *(const v8bf*)(bBase[t] + k0);
      b.h[1] = *(const v8bf*)(bBase[t] + k0 + 16);
      acc[t] = __builtin_amdgcn_wmma_f32_16x16x32_bf16(false, a.v, false, b.v,
                                                       (short)0, acc[t], false, false);
    }
  }
#pragma unroll
  for (int t = 0; t < 4; ++t)
#pragma unroll
    for (int v = 0; v < 8; ++v)
      WX[(size_t)(rt * 16 + v + half * 8) * NCOL + cb0 + t * 16 + l16] = acc[t][v];
}

// ---------------- phase 2: persistent level-scheduled tree scan ----------------
// Block b: column slice s=b%16 (128 cols), replica rep=b/16. B slice lives in LDS
// for the whole kernel. Per round: A) partials for ready nodes, B) combine h/c,
// C) build next ready list. Grid barriers between phases.
__global__ __launch_bounds__(256) void k_scan(const int*   __restrict__ children,
                                              const float* __restrict__ b_iou,
                                              const float* __restrict__ b_f,
                                              const float* __restrict__ WX,
                                              const __bf16* __restrict__ Bt,  // [2048][512] bf16 (U_f | U_iou)
                                              __bf16* __restrict__ Hb,        // [8192,512] bf16 h state
                                              float*  __restrict__ Cst,       // [8192,512] f32 c state
                                              float*  __restrict__ fcBuf,     // [8192,512] f32 partial
                                              float*  __restrict__ iouBuf,    // [8192,1536] f32 partial
                                              int*    __restrict__ sched,     // [bar,gen,cnt0,cnt1], done, list0, list1
                                              float*  __restrict__ out){
  __shared__ __bf16 Bl[128 * BLSTR];   // ~130 KB weight slice, resident whole kernel
  __shared__ __bf16 Al[16 * ALSTR];    // ~16.6 KB child-h staging
  __shared__ int    cidsLds[KCH];

  int* barCnt = sched + 0;
  int* barGen = sched + 1;
  int* lcnt   = sched + 2;       // lcnt[0], lcnt[1]
  int* done   = sched + 8;
  int* lists[2] = { done + NND, done + 2 * NND };

  const int tid  = threadIdx.x;
  const int lane = tid & 31, wv = tid >> 5;     // 8 wave32s
  const int half = lane >> 4, l16 = lane & 15;
  const int s    = blockIdx.x & 15;             // column slice
  const int rep  = blockIdx.x >> 4;             // node replica

  // load B slice into LDS (once), zero A buffer
  for (int ch = tid; ch < 128 * 64; ch += 256){        // 8-elem chunks
    const int col = ch >> 6, k8 = (ch & 63) * 8;
    *(v8bf*)&Bl[col * BLSTR + k8] = *(const v8bf*)&Bt[(size_t)(s * 128 + col) * FD + k8];
  }
  for (int e = tid; e < 16 * ALSTR; e += 256) Al[e] = f2bf(0.f);

  // fixed per-lane bias / column for this wave's tile (indices clamped in-bounds)
  const int myN  = s * 128 + wv * 16 + l16;            // global output col (f view)
  const int myJ  = (s - 4) * 128 + wv * 16 + l16;      // iou index (valid when s>=4)
  const int nSafe = myN & (HD - 1);
  const int jSafe = (myJ >= 0) ? myJ : 0;
  const float bias = (s < 4) ? b_f[nSafe] : b_iou[jSafe];
  const __bf16* blBase = Bl + (size_t)(wv * 16 + l16) * BLSTR + half * 8;
  const __bf16* alBase = Al + (size_t)l16 * ALSTR + half * 8;
  __syncthreads();

  int cur = 0;
  while (true){
    gsync(barCnt, barGen);
    const int R = __hip_atomic_load(&lcnt[cur], __ATOMIC_ACQUIRE, __HIP_MEMORY_SCOPE_AGENT);
    if (R == 0) break;
    const int* list = lists[cur];

    // ---- Phase A: partials for ready nodes ----
    for (int li = rep; li < R; li += REP){
      const int node = list[li];
      if (tid < KCH) cidsLds[tid] = children[node * KCH + tid];
      {
        const int r = tid >> 5;                 // row 0..7
        const int c32 = lane * 16;              // 16 bf16 per thread
        const int cid = children[node * KCH + r];
        v8bf z = {};
        if (cid >= 0){
          const __bf16* src = Hb + (size_t)cid * HD + c32;
          *(v8bf*)&Al[r * ALSTR + c32]     = *(const v8bf*)src;
          *(v8bf*)&Al[r * ALSTR + c32 + 8] = *(const v8bf*)(src + 8);
        } else {
          *(v8bf*)&Al[r * ALSTR + c32]     = z;
          *(v8bf*)&Al[r * ALSTR + c32 + 8] = z;
        }
      }
      __syncthreads();

      v8f acc = (v8f){0,0,0,0,0,0,0,0};
#pragma unroll 4
      for (int ks = 0; ks < 16; ++ks){
        const int k0 = ks * 32;
        BF16x16 a, b;
        a.h[0] = *(const v8bf*)(alBase + k0);
        a.h[1] = *(const v8bf*)(alBase + k0 + 16);
        b.h[0] = *(const v8bf*)(blBase + k0);
        b.h[1] = *(const v8bf*)(blBase + k0 + 16);
        acc = __builtin_amdgcn_wmma_f32_16x16x32_bf16(false, a.v, false, b.v,
                                                      (short)0, acc, false, false);
      }

      if (half == 0){
        if (s < 4){                              // f-gates: rows 0..7 in VGPRs 0..7
          const float pre = WX[(size_t)node * NCOL + myN] + bias;
          float fcv = 0.f;
#pragma unroll
          for (int r = 0; r < KCH; ++r){
            const int cid = cidsLds[r];
            if (cid >= 0)
              fcv += sigm(pre + acc[r]) * Cst[(size_t)cid * HD + myN];
          }
          fcBuf[(size_t)node * HD + myN] = fcv;
        } else {                                 // iou: row-sum of child rows
          float ssum = 0.f;
#pragma unroll
          for (int v = 0; v < 8; ++v) ssum += acc[v];
          iouBuf[(size_t)node * 1536 + myJ] = WX[(size_t)node * NCOL + HD + myJ] + ssum + bias;
        }
      }
      __syncthreads();
    }
    gsync(barCnt, barGen);

    // ---- Phase B: combine gates -> c,h; mark done; reset next count ----
    if (blockIdx.x == 0 && tid == 0)
      __hip_atomic_store(&lcnt[cur ^ 1], 0, __ATOMIC_RELAXED, __HIP_MEMORY_SCOPE_AGENT);
    for (int li = (int)blockIdx.x; li < R; li += G){
      const int node = list[li];
      for (int n = tid; n < HD; n += 256){
        const float ig = sigm(iouBuf[(size_t)node * 1536 + n]);
        const float og = sigm(iouBuf[(size_t)node * 1536 + 512 + n]);
        const float ug = ftanh(iouBuf[(size_t)node * 1536 + 1024 + n]);
        const float c  = ig * ug + fcBuf[(size_t)node * HD + n];
        const float h  = og * ftanh(c);
        Cst[(size_t)node * HD + n] = c;
        Hb[(size_t)node * HD + n]  = f2bf(h);
        out[(size_t)node * HD + n] = h;
      }
      if (tid == 0) done[node] = 1;
    }
    gsync(barCnt, barGen);

    // ---- Phase C: build next ready list ----
    {
      const int i = (int)blockIdx.x * 256 + tid;    // G*256 = 32768 >= NND
      if (i < NND && !done[i]){
        bool ready = true;
#pragma unroll
        for (int r = 0; r < KCH; ++r){
          const int c = children[i * KCH + r];
          if (c >= 0 && !done[c]) ready = false;
        }
        if (ready){
          int idx = __hip_atomic_fetch_add(&lcnt[cur ^ 1], 1, __ATOMIC_RELAXED,
                                           __HIP_MEMORY_SCOPE_AGENT);
          lists[cur ^ 1][idx] = i;
        }
      }
    }
    cur ^= 1;
  }
}

extern "C" void kernel_launch(void* const* d_in, const int* in_sizes, int n_in,
                              void* d_out, int out_size, void* d_ws, size_t ws_size,
                              hipStream_t stream){
  const float* x        = (const float*)d_in[0];
  const int*   children = (const int*)  d_in[1];
  const float* W_iou    = (const float*)d_in[2];
  const float* U_iou    = (const float*)d_in[3];
  const float* b_iou    = (const float*)d_in[4];
  const float* W_f      = (const float*)d_in[5];
  const float* U_f      = (const float*)d_in[6];
  const float* b_f      = (const float*)d_in[7];
  float* out = (float*)d_out;

  char* ws = (char*)d_ws;
  __bf16* Xb     = (__bf16*)ws;  ws += (size_t)NND  * FD * 2;     //  8 MB
  __bf16* Wxt    = (__bf16*)ws;  ws += (size_t)NCOL * FD * 2;     //  2 MB
  __bf16* Bt     = (__bf16*)ws;  ws += (size_t)NCOL * HD * 2;     //  2 MB
  float*  WX     = (float*)ws;   ws += (size_t)NND  * NCOL * 4;   // 64 MB
  __bf16* Hb     = (__bf16*)ws;  ws += (size_t)NND  * HD * 2;     //  8 MB
  float*  Cst    = (float*)ws;   ws += (size_t)NND  * HD * 4;     // 16 MB
  float*  fcBuf  = (float*)ws;   ws += (size_t)NND  * HD * 4;     // 16 MB
  float*  iouBuf = (float*)ws;   ws += (size_t)NND  * 1536 * 4;   // 48 MB
  int*    sched  = (int*)ws;     ws += (size_t)(8 + 3 * NND) * 4; // ~96 KB

  k_cvt_x      <<<(NND * FD + 255) / 256, 256, 0, stream>>>(x, Xb, NND * FD);
  k_prep_w     <<<(NCOL * FD + 255) / 256, 256, 0, stream>>>(W_f, W_iou, Wxt);
  k_prep_w     <<<(NCOL * FD + 255) / 256, 256, 0, stream>>>(U_f, U_iou, Bt);
  k_zero_sched <<<1, 64, 0, stream>>>(sched);
  k_init_leaves<<<(NND + 255) / 256, 256, 0, stream>>>(children, sched);
  k_gemm_wx    <<<2048, 256, 0, stream>>>(Xb, Wxt, WX);
  k_scan       <<<G, 256, 0, stream>>>(children, b_iou, b_f, WX, Bt,
                                       Hb, Cst, fcBuf, iouBuf, sched, out);
}
